// AttributeBranch_69329362092574
// MI455X (gfx1250) — matrix operand order repaired and checked by reading, not compiled, not run on hardware
//
#include <hip/hip_runtime.h>
#include <math.h>

typedef __attribute__((ext_vector_type(16))) _Float16 v16h;
typedef __attribute__((ext_vector_type(8)))  float    v8f;

#define DCH 256          // channel dim
#define HW  3136         // 56*56
#define OCH 32           // conv1 output channels
#define ATTR 400
#define NB  32           // batch
#define PIXB 128         // pixels per workgroup (8 waves x 16)

// ---------------------------------------------------------------------------
// Kernel 1: collapse embedding -> conv -> count-sketch -> circular conv ->
// conv1 weight into a per-batch matrix M[b] in d_ws:  M[b,o,c] (f32, 32x256).
// One block per batch, 256 threads. Tiny (~0.1% of total work).
// ---------------------------------------------------------------------------
__global__ void build_M_kernel(const float* __restrict__ oh,      // [32,400]
                               const float* __restrict__ Wemb,    // [256,400]
                               const float* __restrict__ bemb,    // [256]
                               const float* __restrict__ convw,   // [256,256]
                               const float* __restrict__ convb,   // [256]
                               const float* __restrict__ conv1w,  // [32,256]
                               const int*   __restrict__ h1,
                               const int*   __restrict__ h2,
                               const float* __restrict__ s1,
                               const float* __restrict__ s2,
                               float* __restrict__ Mout)          // [32,32,256]
{
    __shared__ float lds_t[DCH];
    __shared__ float lds_sk1[DCH];
    __shared__ float lds_G[OCH * DCH];     // 32 KB

    const int b   = blockIdx.x;
    const int tid = threadIdx.x;           // 0..255

    // t[c] = attr_one_hot[b] . W_emb[c] + b_emb[c]
    {
        float acc = bemb[tid];
        const float* ohb = oh + b * ATTR;
        const float* wr  = Wemb + tid * ATTR;
        for (int j = 0; j < ATTR; ++j) acc += ohb[j] * wr[j];
        lds_t[tid]   = acc;
        lds_sk1[tid] = 0.f;
    }
    __syncthreads();

    // a[o] = t . conv_w[o] + conv_b[o];  sk1[h1[o]] += s1[o]*a[o]
    {
        float acc = convb[tid];
        const float* wr = convw + tid * DCH;
        for (int c = 0; c < DCH; ++c) acc += lds_t[c] * wr[c];
        atomicAdd(&lds_sk1[h1[tid]], s1[tid] * acc);   // ds_add_f32
    }
    __syncthreads();

    // G[o,e] = sum_c conv1_w[o,(e+c)%256] * sk1[c]   (circular correlation)
    for (int idx = tid; idx < OCH * DCH; idx += 256) {
        const int o = idx >> 8, e = idx & 255;
        const float* wr = conv1w + o * DCH;
        float acc = 0.f;
        for (int c = 0; c < DCH; ++c) acc += wr[(e + c) & 255] * lds_sk1[c];
        lds_G[idx] = acc;
    }
    __syncthreads();

    // M[o,c] = G[o, h2[c]] * s2[c]
    for (int idx = tid; idx < OCH * DCH; idx += 256) {
        const int o = idx >> 8, c = idx & 255;
        Mout[b * (OCH * DCH) + idx] = lds_G[o * DCH + h2[c]] * s2[c];
    }
}

// ---------------------------------------------------------------------------
// Kernel 2 (fused): per-batch GEMM [32x256]x[256x3136] with
// v_wmma_f32_16x16x32_f16 -> bias/relu -> conv2 dot -> sigmoid -> attr_map,
// then (same block, entity slab still cache-hot) attr_feature = map * entity.
// Block = 256 threads = 8 waves; wave w owns pixel tile blockIdx.x*8+w.
// Grid = (ceil(196/8)=25, B=32).
// ---------------------------------------------------------------------------
__global__ void attr_fused_kernel(const float* __restrict__ entity,  // [32,256,3136]
                                  const float* __restrict__ Mws,     // [32,32,256]
                                  const float* __restrict__ conv1b,  // [32]
                                  const float* __restrict__ conv2w,  // [32]
                                  const float* __restrict__ conv2b,  // [1]
                                  float* __restrict__ amap,          // [32,3136]
                                  float* __restrict__ afeat)         // [32,256,3136]
{
    // A fragments pre-swizzled: [rowtile(2)][kblock(8)][lane(32)][elem(16)] f16 = 16 KB
    __shared__ __align__(32) _Float16 ldsA[2 * 8 * 32 * 16];
    __shared__ float ldsAmap[PIXB];

    const int b       = blockIdx.y;
    const int tid     = threadIdx.x;
    const int pixbase = blockIdx.x * PIXB;
    const float* Mb   = Mws + b * (OCH * DCH);

    // Stage M[b] (f32) into LDS as f16 A fragments in this kernel's contiguous k order:
    // lane = 16g+r holds row r, K = 16g+i within a K=32 block.
    for (int e = tid; e < 8192; e += 256) {
        const int i    = e & 15;
        const int lane = (e >> 4) & 31;
        const int kb   = (e >> 9) & 7;
        const int rt   = e >> 12;
        const int g = lane >> 4, r = lane & 15;
        const int kmap = 16 * g + i;
        const int row  = rt * 16 + r;
        const int c    = kb * 32 + kmap;
        ldsA[e] = (_Float16)Mb[row * DCH + c];
    }
    __syncthreads();

    const int wave = tid >> 5;
    const int lane = tid & 31;
    const int tile = blockIdx.x * 8 + wave;     // 16-pixel column tile

    if (tile < HW / 16) {                       // wave-uniform: EXEC all-ones inside
        const int g = lane >> 4, n = lane & 15;
        const int pix = tile * 16 + n;
        const float* entB = entity + (size_t)b * DCH * HW + pix;

        v8f c0 = {};   // rows 0..15
        v8f c1 = {};   // rows 16..31
#pragma unroll 2
        for (int kb = 0; kb < 8; ++kb) {
            // B fragment 32x16: lane 16g+n, elem i -> K = 16g+i, N = n
            v16h bfrag;
            const float* ep = entB + (size_t)(kb * 32 + 16 * g) * HW;
#pragma unroll
            for (int i = 0; i < 16; ++i) {
                const float v = ep[(size_t)i * HW];
                bfrag[i] = (_Float16)fmaxf(v, 0.f);      // fused relu
            }
            const v16h a0 = *(const v16h*)&ldsA[(0 * 8 + kb) * 512 + lane * 16];
            const v16h a1 = *(const v16h*)&ldsA[(1 * 8 + kb) * 512 + lane * 16];
            c0 = __builtin_amdgcn_wmma_f32_16x16x32_f16(false, a0, false, bfrag,
                                                        (short)0, c0, false, false);
            c1 = __builtin_amdgcn_wmma_f32_16x16x32_f16(false, a1, false, bfrag,
                                                        (short)0, c1, false, false);
        }

        // Epilogue: C elem j @ lane 16g+n -> row 8g+j (rowtile1 adds 16), col n.
        float p = 0.f;
#pragma unroll
        for (int j = 0; j < 8; ++j) {
            const int r0 = 8 * g + j;
            const int r1 = 16 + 8 * g + j;
            p += conv2w[r0] * fmaxf(c0[j] + conv1b[r0], 0.f);
            p += conv2w[r1] * fmaxf(c1[j] + conv1b[r1], 0.f);
        }
        p += __shfl_xor(p, 16);                 // combine the two half-wave row groups
        if (g == 0) {
            const float s = 1.f / (1.f + expf(-(p + conv2b[0])));
            amap[b * HW + pix] = s;
            ldsAmap[wave * 16 + n] = s;
        }
    }
    __syncthreads();

    // Fused elementwise: attr_feature = map * entity over this block's
    // 256ch x 128px slab (entity re-read is L0/L2-hot). float4, coalesced.
    const int nvalid = (HW - pixbase < PIXB) ? (HW - pixbase) : PIXB;
    const float* eb = entity + (size_t)b * DCH * HW + pixbase;
    float*       ob = afeat  + (size_t)b * DCH * HW + pixbase;
    for (int idx = tid; idx < DCH * (PIXB / 4); idx += 256) {
        const int c = idx >> 5;                 // channel
        const int p = (idx & 31) * 4;           // pixel offset within slab
        if (p < nvalid) {
            const float4 e = *(const float4*)(eb + (size_t)c * HW + p);
            float4 r;
            r.x = e.x * ldsAmap[p + 0];
            r.y = e.y * ldsAmap[p + 1];
            r.z = e.z * ldsAmap[p + 2];
            r.w = e.w * ldsAmap[p + 3];
            *(float4*)(ob + (size_t)c * HW + p) = r;
        }
    }
}

// ---------------------------------------------------------------------------
extern "C" void kernel_launch(void* const* d_in, const int* in_sizes, int n_in,
                              void* d_out, int out_size, void* d_ws, size_t ws_size,
                              hipStream_t stream) {
    const float* entity  = (const float*)d_in[0];   // [32,256,56,56]
    const float* oh      = (const float*)d_in[1];   // [32,400]
    const float* Wemb    = (const float*)d_in[2];   // [256,400]
    const float* bemb    = (const float*)d_in[3];   // [256]
    const float* convw   = (const float*)d_in[4];   // [256,256]
    const float* convb   = (const float*)d_in[5];   // [256]
    const float* conv1w  = (const float*)d_in[6];   // [32,256]
    const float* conv1b  = (const float*)d_in[7];   // [32]
    const float* conv2w  = (const float*)d_in[8];   // [1,32]
    const float* conv2b  = (const float*)d_in[9];   // [1]
    const int*   h1      = (const int*)d_in[10];
    const int*   h2      = (const int*)d_in[11];
    const float* s1      = (const float*)d_in[12];
    const float* s2      = (const float*)d_in[13];

    float* amap  = (float*)d_out;                   // [32,3136] = 100352
    float* afeat = (float*)d_out + NB * HW;         // [32,256,3136]
    float* Mws   = (float*)d_ws;                    // 32*32*256 f32 = 1 MB

    build_M_kernel<<<NB, 256, 0, stream>>>(oh, Wemb, bemb, convw, convb,
                                           conv1w, h1, h2, s1, s2, Mws);

    attr_fused_kernel<<<dim3((HW / 16 + 7) / 8, NB), 256, 0, stream>>>(
        entity, Mws, conv1b, conv2w, conv2b, amap, afeat);
}
